// MultiheadAttention_29240137351465
// MI455X (gfx1250) — compile-verified, matrix operand order checked by reading, not run on hardware
//
#include <hip/hip_runtime.h>
#include <hip/hip_bf16.h>

// ---------------------------------------------------------------------------
// Multihead attention forward for MI455X (gfx1250, wave32, WMMA f16->f32).
//  B=4, S=2048, D=1024, H=16, Dh=64.
//    1) proj_f16_kernel x3 : Q/K/V = X @ W^T (f32 in, f16 out; flat layout ==
//       [B,H,S,Dh] because the reference uses a *direct* reshape)
//    2) attn_kernel        : streaming masked exp-softmax attention; K tile is
//       staged into LDS by the Tensor Data Mover (tensor_load_to_lds, with
//       hardware row padding), V is transpose-staged manually.
//    3) fc_kernel          : outs = res^T @ fc_w^T + fc_b (fp32 out)
// ---------------------------------------------------------------------------

typedef __attribute__((ext_vector_type(16))) _Float16 v16h;
typedef __attribute__((ext_vector_type(8)))  _Float16 v8h;
typedef __attribute__((ext_vector_type(8)))  float    v8f;
typedef unsigned int u32;
typedef __attribute__((ext_vector_type(4)))  u32 v4u;
typedef __attribute__((ext_vector_type(8)))  int v8i;
typedef __attribute__((ext_vector_type(4)))  int v4i;
typedef __attribute__((address_space(3)))    _Float16 lds_h;

#ifndef USE_TDM
#if defined(__has_builtin)
#if __has_builtin(__builtin_amdgcn_tensor_load_to_lds) && \
    __has_builtin(__builtin_amdgcn_s_wait_tensorcnt)
#define USE_TDM 1
#endif
#endif
#endif
#ifndef USE_TDM
#define USE_TDM 0
#endif

__device__ __forceinline__ v8f wmma_f16(v16h a, v16h b, v8f c) {
  // D = A(16x32,f16) x B(32x16,f16) + C(16x16,f32)
  return __builtin_amdgcn_wmma_f32_16x16x32_f16(
      /*neg_a=*/false, a, /*neg_b=*/false, b,
      /*c_mod=*/(short)0, c, /*reuse_a=*/false, /*reuse_b=*/false);
}

// Gather one 16-half operand fragment from LDS.
// CDNA5 16-bit A layout: halves 0..7 hold K = kbase..kbase+7, halves 8..15
// hold K = kbase+16..kbase+23 (kbase = 0 for lanes 0-15, 8 for lanes 16-31).
// `p` already points at row_base + kbase; both chunks are 16B aligned.
__device__ __forceinline__ v16h ldfrag16(const _Float16* p) {
  v8h lo = *(const v8h*)(p);
  v8h hi = *(const v8h*)(p + 16);
  v16h r;
#pragma unroll
  for (int i = 0; i < 8; ++i) { r[i] = lo[i]; r[i + 8] = hi[i]; }
  return r;
}

__device__ __forceinline__ void st4h(_Float16* d, float4 f) {
  d[0] = (_Float16)f.x; d[1] = (_Float16)f.y;
  d[2] = (_Float16)f.z; d[3] = (_Float16)f.w;
}

#if USE_TDM
// Issue one TDM 2D tile load (f16 elements): tile_h rows x tile_w elements,
// row stride in memory = stride_e elements, LDS rows padded by pad_amt_code
// (+1 DWORDs) after every (2<<pad_int_code) DWORDs. All inputs must be
// wave-uniform (descriptors live in SGPRs).
__device__ __forceinline__ void tdm_load_2d_f16(u32 lds_byte, unsigned long long ga,
                                                u32 tile_w, u32 tile_h, u32 stride_e,
                                                u32 pad_int_code, u32 pad_amt_code) {
  v4u g0;
  g0[0] = 1u;                                          // count=1 (valid), user mode
  g0[1] = lds_byte;                                    // lds_addr  [63:32]
  g0[2] = (u32)(ga & 0xffffffffull);                   // global_addr lo
  g0[3] = (u32)((ga >> 32) & 0x01ffffffull)            // global_addr [56:32]
        | (2u << 30);                                  // type = 2 ("image")
  v8i g1;
  g1[0] = (int)((1u << 16)                             // data_size = 2 bytes
              | (1u << 20)                             // pad_enable
              | (pad_int_code << 22)                   // pad_interval
              | (pad_amt_code << 25));                 // pad_amount
  g1[1] = (int)(tile_w << 16);                         // tensor_dim0 low16
  g1[2] = (int)(tile_h << 16);                         // tensor_dim1 low16
  g1[3] = (int)(tile_w << 16);                         // tile_dim0
  g1[4] = (int)tile_h;                                 // tile_dim1 (tile_dim2=0)
  g1[5] = (int)stride_e;                               // tensor_dim0_stride low32
  g1[6] = 0;
  g1[7] = 0;
  v4i z4 = {0, 0, 0, 0};
#if __clang_major__ >= 23
  v8i z8 = {0, 0, 0, 0, 0, 0, 0, 0};
  __builtin_amdgcn_tensor_load_to_lds(g0, g1, z4, z4, z8, 0);
#else
  __builtin_amdgcn_tensor_load_to_lds(g0, g1, z4, z4, 0);
#endif
}
#endif

// ---------------------------------------------------------------------------
// Kernel 1: C[m,n] = sum_k X[m,k] * W[n,k],  M=8192 N=1024 K=1024, f16 output.
// Block tile 128(M) x 64(N), K-step 32. 8 waves, each computes 32x32.
// ---------------------------------------------------------------------------
__global__ __launch_bounds__(256, 1)
void proj_f16_kernel(const float* __restrict__ X, const float* __restrict__ W,
                     _Float16* __restrict__ Out) {
  __shared__ _Float16 Al[128 * 40];   // 128x32, stride 40 halfs (80B, 16B-aligned)
  __shared__ _Float16 Bl[64 * 40];    // 64x32  (W rows = output columns)

  const int tid   = threadIdx.x;
  const int lane  = tid & 31;
  const int wave  = tid >> 5;
  const int wm    = wave >> 1;
  const int wn    = wave & 1;
  const int row16 = lane & 15;
  const int hi    = lane >> 4;
  const int kbase = hi ? 8 : 0;
  const int m0 = blockIdx.x * 128;
  const int n0 = blockIdx.y * 64;

  v8f acc00 = {}, acc01 = {}, acc10 = {}, acc11 = {};

  for (int k0 = 0; k0 < 1024; k0 += 32) {
    __syncthreads();
#pragma unroll
    for (int i = 0; i < 4; ++i) {         // A tile: 128x32 f32 -> f16
      int c = tid + i * 256; int r = c >> 3; int col = (c & 7) * 4;
      float4 f = *(const float4*)(X + (size_t)(m0 + r) * 1024 + k0 + col);
      st4h(&Al[r * 40 + col], f);
    }
#pragma unroll
    for (int i = 0; i < 2; ++i) {         // W tile: 64x32 f32 -> f16
      int c = tid + i * 256; int r = c >> 3; int col = (c & 7) * 4;
      float4 f = *(const float4*)(W + (size_t)(n0 + r) * 1024 + k0 + col);
      st4h(&Bl[r * 40 + col], f);
    }
    __syncthreads();

    v16h a0 = ldfrag16(&Al[(wm * 32 + row16) * 40 + kbase]);
    v16h a1 = ldfrag16(&Al[(wm * 32 + 16 + row16) * 40 + kbase]);
    v16h b0 = ldfrag16(&Bl[(wn * 32 + row16) * 40 + kbase]);
    v16h b1 = ldfrag16(&Bl[(wn * 32 + 16 + row16) * 40 + kbase]);
    acc00 = wmma_f16(a0, b0, acc00);
    acc01 = wmma_f16(a0, b1, acc01);
    acc10 = wmma_f16(a1, b0, acc10);
    acc11 = wmma_f16(a1, b1, acc11);
  }

  // C layout: VGPR r -> row (r | r+8 via lane hi), lanes 0..15 -> column.
  // Direct-reshape trick: flat [m,n] IS the [B,H,S,Dh] head-major layout.
#pragma unroll
  for (int r = 0; r < 8; ++r) {
    int mr = m0 + wm * 32 + r + 8 * hi;
    int nc = n0 + wn * 32 + row16;
    Out[(size_t)mr * 1024 + nc]             = (_Float16)acc00[r];
    Out[(size_t)mr * 1024 + nc + 16]        = (_Float16)acc01[r];
    Out[(size_t)(mr + 16) * 1024 + nc]      = (_Float16)acc10[r];
    Out[(size_t)(mr + 16) * 1024 + nc + 16] = (_Float16)acc11[r];
  }
}

// ---------------------------------------------------------------------------
// Kernel 2: streaming masked-exp attention. One block = 128 query rows of one
// (b,h); wave w owns rows [w*16, w*16+16). Keys/values streamed 32 at a time.
// ---------------------------------------------------------------------------
__global__ __launch_bounds__(256, 1)
void attn_kernel(const _Float16* __restrict__ Qh, const _Float16* __restrict__ Kh,
                 const _Float16* __restrict__ Vh, const unsigned char* __restrict__ mask,
                 float* __restrict__ outRes, _Float16* __restrict__ resH) {
  const int S = 2048, Dh = 64, B = 4;
  __shared__ _Float16 Ql[128 * 72];   // 128 x 64 query tile (stride 72 halfs)
  __shared__ _Float16 Kl[32 * 72];    // 32  x 64 key tile (row = key index)
  __shared__ _Float16 Vt[64 * 40];    // 64(dh) x 32(k): V transposed for B frags
  __shared__ _Float16 Pl[8][16 * 40]; // per-wave 16 x 32 probability tile

  const int tid   = threadIdx.x;
  const int lane  = tid & 31;
  const int wave  = tid >> 5;
  const int row16 = lane & 15;
  const int hi    = lane >> 4;
  const int kbase = hi ? 8 : 0;
  const int qb = blockIdx.x * 128;
  const int bh = blockIdx.y;            // b*16 + h
  const int b  = bh >> 4;
  const int h  = bh & 15;

  const _Float16* Qp = Qh + (size_t)bh * S * Dh;
  const _Float16* Kp = Kh + (size_t)bh * S * Dh;
  const _Float16* Vp = Vh + (size_t)bh * S * Dh;
  const unsigned char* mb = mask + (size_t)b * S * S;

#if USE_TDM
  const int wu = __builtin_amdgcn_readfirstlane(wave);  // wave-uniform wave id
  const u32 kl_base = (u32)(unsigned long long)(lds_h*)&Kl[0];
#endif

  // Stage the 128x64 Q tile once.
#pragma unroll
  for (int i = 0; i < 4; ++i) {
    int c = tid + i * 256; int r = c >> 3; int col = (c & 7) * 8;
    *(v8h*)(&Ql[r * 72 + col]) = *(const v8h*)(Qp + (size_t)(qb + r) * Dh + col);
  }
  __syncthreads();
  const v16h aq0 = ldfrag16(&Ql[(wave * 16 + row16) * 72 + 0  + kbase]);
  const v16h aq1 = ldfrag16(&Ql[(wave * 16 + row16) * 72 + 32 + kbase]);

  v8f acc0 = {}, acc1 = {}, acc2 = {}, acc3 = {};
  float dpart[8];
#pragma unroll
  for (int r = 0; r < 8; ++r) dpart[r] = 0.f;

  for (int kb = 0; kb < S; kb += 32) {
    __syncthreads();
#if USE_TDM
    // Tensor Data Mover: each wave DMAs 4 key rows (4 x 64 f16) into Kl with
    // hardware row padding: pad_interval code 4 (= 32 DWORDs = one 128B row),
    // pad_amount code 3 (= 4 DWORDs = 8 halfs) -> LDS row stride 144B.
    tdm_load_2d_f16(kl_base + (u32)(wu * 4 * 144),
                    (unsigned long long)(const void*)(Kp + (size_t)(kb + wu * 4) * Dh),
                    /*tile_w=*/64, /*tile_h=*/4, /*stride_e=*/64,
                    /*pad_int=*/4, /*pad_amt=*/3);
#endif
    {  // stage transposed V tile (each thread: one 8-half chunk)
      int r = tid >> 3, col = (tid & 7) * 8;
#if !USE_TDM
      *(v8h*)(&Kl[r * 72 + col]) = *(const v8h*)(Kp + (size_t)(kb + r) * Dh + col);
#endif
      v8h vv = *(const v8h*)(Vp + (size_t)(kb + r) * Dh + col);
#pragma unroll
      for (int j = 0; j < 8; ++j) Vt[(col + j) * 40 + r] = vv[j];
    }
    if (kb + 32 < S) {  // uniform branch: EXEC stays all-ones
      __builtin_prefetch(Kp + (size_t)(kb + 32) * Dh + lane * 16, 0, 0);
      __builtin_prefetch(Vp + (size_t)(kb + 32) * Dh + lane * 16, 0, 0);
    }
#if USE_TDM
    __builtin_amdgcn_s_wait_tensorcnt(0);  // this wave's DMA slice done
#endif
    __syncthreads();                       // all slices published

    _Float16* Pw = &Pl[wave][0];
#pragma unroll
    for (int kt = 0; kt < 2; ++kt) {  // logits 16(q) x 16(key), K-dim = 64
      v8f c8 = {};
      v16h bk0 = ldfrag16(&Kl[(kt * 16 + row16) * 72 + 0  + kbase]);
      c8 = wmma_f16(aq0, bk0, c8);
      v16h bk1 = ldfrag16(&Kl[(kt * 16 + row16) * 72 + 32 + kbase]);
      c8 = wmma_f16(aq1, bk1, c8);

      const int kcol = kb + kt * 16 + row16;  // this lane's key column
#pragma unroll
      for (int r = 0; r < 8; ++r) {
        int mrow = r + 8 * hi;                         // 0..15 within wave tile
        int q    = qb + wave * 16 + mrow;
        float e  = __expf(c8[r] * 0.125f);             // scale = 1/sqrt(64)
        float p  = mb[(size_t)q * S + kcol] ? e : 0.f; // masked_fill(False->0)
        dpart[r] += p;
        Pw[mrow * 40 + kt * 16 + row16] = (_Float16)p;
      }
    }
    // Same-wave LDS turnaround (cross-lane): LDS is in-order per wave.
    asm volatile("s_wait_dscnt 0" ::: "memory");

    v16h ap  = ldfrag16(&Pw[row16 * 40 + kbase]);      // P as A operand (16x32)
    v16h bv0 = ldfrag16(&Vt[(0 * 16 + row16) * 40 + kbase]);
    acc0 = wmma_f16(ap, bv0, acc0);
    v16h bv1 = ldfrag16(&Vt[(1 * 16 + row16) * 40 + kbase]);
    acc1 = wmma_f16(ap, bv1, acc1);
    v16h bv2 = ldfrag16(&Vt[(2 * 16 + row16) * 40 + kbase]);
    acc2 = wmma_f16(ap, bv2, acc2);
    v16h bv3 = ldfrag16(&Vt[(3 * 16 + row16) * 40 + kbase]);
    acc3 = wmma_f16(ap, bv3, acc3);
  }

  // Row denominators: each half-wave's 16 lanes hold disjoint key-column partials.
  float den[8];
#pragma unroll
  for (int r = 0; r < 8; ++r) {
    float v = dpart[r];
    v += __shfl_xor(v, 1, 16);
    v += __shfl_xor(v, 2, 16);
    v += __shfl_xor(v, 4, 16);
    v += __shfl_xor(v, 8, 16);
    den[r] = v;
  }

#pragma unroll
  for (int r = 0; r < 8; ++r) {
    int q = qb + wave * 16 + r + 8 * hi;
    float d = den[r];
    float inv = (d > 0.f) ? (1.f / d) : 0.f;           // all-masked row -> 0
    float v0 = acc0[r] * inv, v1 = acc1[r] * inv;
    float v2 = acc2[r] * inv, v3 = acc3[r] * inv;
    size_t ro = (((size_t)h * B + b) * S + q) * Dh;    // res.transpose(1,0,2,3)
    outRes[ro + 0  + row16] = v0;
    outRes[ro + 16 + row16] = v1;
    outRes[ro + 32 + row16] = v2;
    outRes[ro + 48 + row16] = v3;
    size_t fo = ((size_t)(b * S + q)) * 1024 + h * Dh; // [B,S,D] f16 for the FC
    resH[fo + 0  + row16] = (_Float16)v0;
    resH[fo + 16 + row16] = (_Float16)v1;
    resH[fo + 32 + row16] = (_Float16)v2;
    resH[fo + 48 + row16] = (_Float16)v3;
  }
}

// ---------------------------------------------------------------------------
// Kernel 3: outs[m,n] = sum_k res^T[m,k]*fc_w[n,k] + fc_b[n]  (fp32 out)
// ---------------------------------------------------------------------------
__global__ __launch_bounds__(256, 1)
void fc_kernel(const _Float16* __restrict__ A, const float* __restrict__ W,
               const float* __restrict__ bias, float* __restrict__ Out) {
  __shared__ _Float16 Al[128 * 40];
  __shared__ _Float16 Bl[64 * 40];

  const int tid   = threadIdx.x;
  const int lane  = tid & 31;
  const int wave  = tid >> 5;
  const int wm    = wave >> 1;
  const int wn    = wave & 1;
  const int row16 = lane & 15;
  const int hi    = lane >> 4;
  const int kbase = hi ? 8 : 0;
  const int m0 = blockIdx.x * 128;
  const int n0 = blockIdx.y * 64;

  v8f acc00 = {}, acc01 = {}, acc10 = {}, acc11 = {};

  for (int k0 = 0; k0 < 1024; k0 += 32) {
    __syncthreads();
#pragma unroll
    for (int i = 0; i < 2; ++i) {         // A tile (already f16): 128x32
      int c = tid + i * 256; int r = c >> 2; int col = (c & 3) * 8;
      *(v8h*)(&Al[r * 40 + col]) = *(const v8h*)(A + (size_t)(m0 + r) * 1024 + k0 + col);
    }
#pragma unroll
    for (int i = 0; i < 2; ++i) {         // fc_w tile f32 -> f16
      int c = tid + i * 256; int r = c >> 3; int col = (c & 7) * 4;
      float4 f = *(const float4*)(W + (size_t)(n0 + r) * 1024 + k0 + col);
      st4h(&Bl[r * 40 + col], f);
    }
    __syncthreads();

    v16h a0 = ldfrag16(&Al[(wm * 32 + row16) * 40 + kbase]);
    v16h a1 = ldfrag16(&Al[(wm * 32 + 16 + row16) * 40 + kbase]);
    v16h b0 = ldfrag16(&Bl[(wn * 32 + row16) * 40 + kbase]);
    v16h b1 = ldfrag16(&Bl[(wn * 32 + 16 + row16) * 40 + kbase]);
    acc00 = wmma_f16(a0, b0, acc00);
    acc01 = wmma_f16(a0, b1, acc01);
    acc10 = wmma_f16(a1, b0, acc10);
    acc11 = wmma_f16(a1, b1, acc11);
  }

#pragma unroll
  for (int r = 0; r < 8; ++r) {
    int mr = m0 + wm * 32 + r + 8 * hi;
    int nc = n0 + wn * 32 + row16;
    Out[(size_t)mr * 1024 + nc]             = acc00[r] + bias[nc];
    Out[(size_t)mr * 1024 + nc + 16]        = acc01[r] + bias[nc + 16];
    Out[(size_t)(mr + 16) * 1024 + nc]      = acc10[r] + bias[nc];
    Out[(size_t)(mr + 16) * 1024 + nc + 16] = acc11[r] + bias[nc + 16];
  }
}

// ---------------------------------------------------------------------------
extern "C" void kernel_launch(void* const* d_in, const int* in_sizes, int n_in,
                              void* d_out, int out_size, void* d_ws, size_t ws_size,
                              hipStream_t stream) {
  (void)in_sizes; (void)n_in; (void)out_size; (void)ws_size;

  const float* query = (const float*)d_in[0];
  const float* key_t = (const float*)d_in[1];
  const float* value = (const float*)d_in[2];
  const unsigned char* mask = (const unsigned char*)d_in[3];  // jax bool: 1 byte
  const float* Wq   = (const float*)d_in[4];
  const float* Wk   = (const float*)d_in[5];
  const float* Wv   = (const float*)d_in[6];
  const float* fc_w = (const float*)d_in[7];
  const float* fc_b = (const float*)d_in[8];

  float* outs   = (float*)d_out;                       // [B,S,D]    8388608 f32
  float* outRes = outs + (size_t)4 * 2048 * 1024;      // [H,B,S,Dh] 8388608 f32

  // Workspace: 4 x 16MiB f16 buffers (Q, K, V head-major; res [B,S,D]).
  char* ws = (char*)d_ws;
  const size_t FBUF = (size_t)8192 * 1024 * sizeof(_Float16);
  _Float16* qf = (_Float16*)(ws);
  _Float16* kf = (_Float16*)(ws + FBUF);
  _Float16* vf = (_Float16*)(ws + 2 * FBUF);
  _Float16* rf = (_Float16*)(ws + 3 * FBUF);

  dim3 blk(256);
  dim3 gproj(64, 16);   // M/128 x N/64
  proj_f16_kernel<<<gproj, blk, 0, stream>>>(query, Wq, qf);
  proj_f16_kernel<<<gproj, blk, 0, stream>>>(key_t, Wk, kf);
  proj_f16_kernel<<<gproj, blk, 0, stream>>>(value, Wv, vf);

  dim3 gattn(16, 64);   // S/128 query blocks x (B*H)
  attn_kernel<<<gattn, blk, 0, stream>>>(qf, kf, vf, mask, outRes, rf);

  dim3 gfc(64, 16);
  fc_kernel<<<gfc, blk, 0, stream>>>(rf, fc_w, fc_b, outs);
}
